// SpatViT_72765335928971
// MI455X (gfx1250) — compile-verified, matrix operand order checked by reading, not run on hardware
//
#include <hip/hip_runtime.h>
#include <math.h>
#include <stdint.h>

// ---------------- types ----------------
typedef __attribute__((ext_vector_type(16))) __bf16 v16bf;
typedef __attribute__((ext_vector_type(8)))  float  v8f;

static __device__ __forceinline__ v8f wmma_bf16(v16bf a, v16bf b, v8f c) {
    return __builtin_amdgcn_wmma_f32_16x16x32_bf16(
        /*neg_a=*/false, a, /*neg_b=*/false, b,
        /*c_mod=*/(short)0, c, /*reuse_a=*/false, /*reuse_b=*/false);
}

// per-lane 16B global->LDS async copy (CDNA5, ASYNCcnt-tracked)
static __device__ __forceinline__ void async_ld_lds_b128(uint32_t lds_off, const void* gptr) {
    asm volatile("global_load_async_to_lds_b128 %0, %1, off"
                 :: "v"(lds_off), "v"(gptr) : "memory");
}
static __device__ __forceinline__ void wait_async0() {
    asm volatile("s_wait_asynccnt 0x0" ::: "memory");
}

#define BM 128
#define BN 128
#define KC 32

// ---------------- bf16 GEMM: C[M,N] = A[M,K] x W[Npad,K]^T (+bias)(+C) --------
// A, W bf16; C fp32. W is zero-padded to Npad = gridDim.y*BN rows.
// flags bit0: add bias, bit1: accumulate into existing C.
// Double-buffered async-to-LDS staging.
__global__ __launch_bounds__(256)
void gemm_bf16_kernel(const __bf16* __restrict__ A, int lda,
                      const __bf16* __restrict__ W,
                      const float* __restrict__ bias,
                      float* __restrict__ C, int ldc,
                      int M, int N, int K, int flags)
{
    __shared__ __bf16 sA[2][BM][KC];   // 2 x 8 KB
    __shared__ __bf16 sB[2][BN][KC];   // 2 x 8 KB

    const int tid  = threadIdx.x;
    const int lane = tid & 31;
    const int wid  = tid >> 5;       // 0..7
    const int wm   = wid >> 1;       // 0..3 : 32-row stripe
    const int wn   = wid & 1;        // 0..1 : 64-col stripe
    const int m0   = blockIdx.x * BM;
    const int n0   = blockIdx.y * BN;

    const v8f zf = {0.f,0.f,0.f,0.f,0.f,0.f,0.f,0.f};
    v8f acc[2][4] = {{zf,zf,zf,zf},{zf,zf,zf,zf}};

    // staging: each thread owns two 16B segments per tile (rows r0 and r0+64)
    const int r0   = tid >> 2;         // 0..63
    const int cs   = (tid & 3) * 8;    // bf16 col: 0,8,16,24

    auto stage = [&](int buf, int k0) {
        #pragma unroll
        for (int it = 0; it < 2; it++) {
            const int row = r0 + it * 64;
            async_ld_lds_b128((uint32_t)(uintptr_t)&sA[buf][row][cs],
                              (const void*)(A + (size_t)(m0 + row) * lda + k0 + cs));
            async_ld_lds_b128((uint32_t)(uintptr_t)&sB[buf][row][cs],
                              (const void*)(W + (size_t)(n0 + row) * K + k0 + cs));
        }
    };

    const int nk = K / KC;
    stage(0, 0);
    wait_async0();
    __syncthreads();

    const int fr  = lane & 15;
    const int fko = (lane >> 4) * 16;

    for (int kc = 0; kc < nk; kc++) {
        const int cur = kc & 1;
        if (kc + 1 < nk) stage(cur ^ 1, (kc + 1) * KC);

        v16bf a0 = *(const v16bf*)&sA[cur][wm * 32 + fr][fko];
        v16bf a1 = *(const v16bf*)&sA[cur][wm * 32 + 16 + fr][fko];
        #pragma unroll
        for (int j = 0; j < 4; j++) {
            v16bf bj = *(const v16bf*)&sB[cur][wn * 64 + j * 16 + fr][fko];
            acc[0][j] = wmma_bf16(a0, bj, acc[0][j]);
            acc[1][j] = wmma_bf16(a1, bj, acc[1][j]);
        }
        wait_async0();
        __syncthreads();
    }

    // epilogue: C layout -> lane l holds col (l&15), rows r + 8*(l>=16)
    const int nlane = lane & 15;
    const int moff  = (lane >> 4) * 8;
    #pragma unroll
    for (int i = 0; i < 2; i++) {
        #pragma unroll
        for (int j = 0; j < 4; j++) {
            const int col = n0 + wn * 64 + j * 16 + nlane;
            if (col >= N) continue;
            const float bv = (flags & 1) ? bias[col] : 0.f;
            #pragma unroll
            for (int r = 0; r < 8; r++) {
                const int row = m0 + wm * 32 + i * 16 + moff + r;
                float v = acc[i][j][r] + bv;
                if (flags & 2) v += C[(size_t)row * ldc + col];
                C[(size_t)row * ldc + col] = v;
            }
        }
    }
}

// ---------------- fp32 -> bf16 convert (strided src, zero-padded rows) --------
__global__ void cvt_bf16_kernel(const float* __restrict__ src, int ld_src,
                                __bf16* __restrict__ dst,
                                int rows_pad, int rows_valid, int cols)
{
    const int idx = blockIdx.x * blockDim.x + threadIdx.x;
    if (idx >= rows_pad * cols) return;
    const int row = idx / cols;
    const int col = idx - row * cols;
    dst[idx] = (row < rows_valid) ? (__bf16)src[(size_t)row * ld_src + col] : (__bf16)0.f;
}

// ---------------- im2col for patch embed (bf16 out) ----------------
__global__ void im2col_kernel(const float* __restrict__ x, __bf16* __restrict__ A)
{
    const int idx = blockIdx.x * blockDim.x + threadIdx.x;
    if (idx >= 2048 * 768) return;
    const int col = idx % 768;
    const int row = idx / 768;
    const int b  = row >> 10;
    const int t  = row & 1023;
    const int hy = t >> 5, wx = t & 31;
    const int c  = col >> 8;
    const int pq = col & 255;
    const int p  = pq >> 4, q = pq & 15;
    A[idx] = (__bf16)x[(((size_t)b * 3 + c) * 512 + (hy * 16 + p)) * 512 + (wx * 16 + q)];
}

// ---------------- LayerNorm: wave per row of 768, bf16 out ----------------
__global__ __launch_bounds__(256)
void ln_kernel(const float* __restrict__ h, const float* __restrict__ w,
               const float* __restrict__ b, __bf16* __restrict__ out, int rows)
{
    const int lane = threadIdx.x & 31;
    const int row  = blockIdx.x * 8 + (threadIdx.x >> 5);
    if (row >= rows) return;
    const float* x = h + (size_t)row * 768;
    float v[24];
    float s = 0.f;
    #pragma unroll
    for (int i = 0; i < 24; i++) { v[i] = x[lane + i * 32]; s += v[i]; }
    #pragma unroll
    for (int m = 16; m >= 1; m >>= 1) s += __shfl_xor(s, m, 32);
    const float mean = s * (1.f / 768.f);
    float var = 0.f;
    #pragma unroll
    for (int i = 0; i < 24; i++) { float d = v[i] - mean; var += d * d; }
    #pragma unroll
    for (int m = 16; m >= 1; m >>= 1) var += __shfl_xor(var, m, 32);
    const float rstd = rsqrtf(var * (1.f / 768.f) + 1e-6f);
    __bf16* o = out + (size_t)row * 768;
    #pragma unroll
    for (int i = 0; i < 24; i++) {
        const int c = lane + i * 32;
        o[c] = (__bf16)((v[i] - mean) * rstd * w[c] + b[c]);
    }
}

// ---------------- exact GELU (fp32 in, bf16 out) ----------------
__global__ void gelu_kernel(const float* __restrict__ x, __bf16* __restrict__ y, int n)
{
    const int i = blockIdx.x * blockDim.x + threadIdx.x;
    if (i < n) {
        const float v = x[i];
        y[i] = (__bf16)(0.5f * v * (1.f + erff(v * 0.70710678118654752f)));
    }
}

// ---------------- flash-style dense attention ----------------
// qkv [B,1024,2304] fp32; ao [B,1024,768] bf16. grid (B*H, N/64), 128 threads.
__global__ __launch_bounds__(128)
void flash_attn_kernel(const float* __restrict__ qkv, __bf16* __restrict__ ao)
{
    __shared__ __bf16 kbuf[32][64];     // [key][dim]  4 KB
    __shared__ __bf16 vbuf[64][32];     // [dim][key]  4 KB (transposed for PV)
    __shared__ __bf16 pbuf[4][16][32];  // per-wave P  4 KB

    const int tid  = threadIdx.x;
    const int lane = tid & 31;
    const int w    = tid >> 5;                 // 0..3
    const int bh   = blockIdx.x;               // 0..23
    const int b    = bh / 12, hh = bh % 12;
    const int qbase = blockIdx.y * 64 + w * 16;

    const int fr   = lane & 15;
    const int fko  = (lane >> 4) * 16;
    const int moff = (lane >> 4) * 8;

    // Q fragments (scale folded in)
    v16bf aq[2];
    {
        const float* qrow = qkv + ((size_t)b * 1024 + qbase + fr) * 2304 + hh * 64;
        __bf16 t[16] __attribute__((aligned(32)));
        #pragma unroll
        for (int c = 0; c < 2; c++) {
            #pragma unroll
            for (int u = 0; u < 16; u++) t[u] = (__bf16)(qrow[c * 32 + fko + u] * 0.125f);
            aq[c] = *(v16bf*)t;
        }
    }

    const v8f zf = {0.f,0.f,0.f,0.f,0.f,0.f,0.f,0.f};
    v8f o[4] = {zf, zf, zf, zf};
    float mrow[8], lrow[8];
    #pragma unroll
    for (int r = 0; r < 8; r++) { mrow[r] = -1e30f; lrow[r] = 0.f; }

    for (int j = 0; j < 1024; j += 32) {
        __syncthreads();
        // stage K,V chunk (32 keys); fp32 -> bf16, V transposed
        {
            const int key = tid >> 2;           // 0..31
            const int dg  = (tid & 3) * 16;     // 0,16,32,48
            const float* kr = qkv + ((size_t)b * 1024 + j + key) * 2304 + 768 + hh * 64 + dg;
            const float* vr = kr + 768;
            __bf16 t[16] __attribute__((aligned(32)));
            #pragma unroll
            for (int u = 0; u < 16; u++) t[u] = (__bf16)kr[u];
            *(v16bf*)&kbuf[key][dg] = *(v16bf*)t;
            #pragma unroll
            for (int u = 0; u < 16; u++) vbuf[dg + u][key] = (__bf16)vr[u];
        }
        __syncthreads();

        // S = (Q*scale) K^T over two 16-key groups
        v8f s[2];
        #pragma unroll
        for (int g = 0; g < 2; g++) {
            v16bf b0 = *(const v16bf*)&kbuf[g * 16 + fr][fko];
            v16bf b1 = *(const v16bf*)&kbuf[g * 16 + fr][32 + fko];
            v8f z = zf;
            z    = wmma_bf16(aq[0], b0, z);
            s[g] = wmma_bf16(aq[1], b1, z);
        }

        // online softmax (row reductions across the 16-lane group)
        float p0v[8], p1v[8];
        #pragma unroll
        for (int r = 0; r < 8; r++) {
            float cm = fmaxf(s[0][r], s[1][r]);
            #pragma unroll
            for (int m = 1; m < 16; m <<= 1) cm = fmaxf(cm, __shfl_xor(cm, m, 32));
            const float nm = fmaxf(mrow[r], cm);
            const float alpha = __expf(mrow[r] - nm);
            const float p0 = __expf(s[0][r] - nm);
            const float p1 = __expf(s[1][r] - nm);
            float rs = p0 + p1;
            #pragma unroll
            for (int m = 1; m < 16; m <<= 1) rs += __shfl_xor(rs, m, 32);
            lrow[r] = lrow[r] * alpha + rs;
            mrow[r] = nm;
            p0v[r] = p0; p1v[r] = p1;
            #pragma unroll
            for (int t = 0; t < 4; t++) o[t][r] *= alpha;
        }

        // P -> LDS -> A fragment
        #pragma unroll
        for (int r = 0; r < 8; r++) {
            pbuf[w][moff + r][fr]      = (__bf16)p0v[r];
            pbuf[w][moff + r][16 + fr] = (__bf16)p1v[r];
        }
        __syncthreads();
        v16bf ap = *(const v16bf*)&pbuf[w][fr][fko];

        // O += P V  (4 feature tiles of 16)
        #pragma unroll
        for (int t = 0; t < 4; t++) {
            v16bf bv = *(const v16bf*)&vbuf[t * 16 + fr][fko];
            o[t] = wmma_bf16(ap, bv, o[t]);
        }
    }

    // normalize and store (bf16)
    #pragma unroll
    for (int t = 0; t < 4; t++) {
        #pragma unroll
        for (int r = 0; r < 8; r++) {
            const int tokq = qbase + moff + r;
            ao[((size_t)b * 1024 + tokq) * 768 + hh * 64 + t * 16 + fr] =
                (__bf16)(o[t][r] / lrow[r]);
        }
    }
}

// ---------------- deformable attention (sample layers) ----------------
// One wave per (b,h,token). qkv [B,1024,2304] fp32, off [B,1024,96] fp32,
// ao [B,1024,768] bf16.
__global__ __launch_bounds__(256)
void deform_attn_kernel(const float* __restrict__ qkv, const float* __restrict__ off,
                        __bf16* __restrict__ ao)
{
    const int lane = threadIdx.x & 31;
    const int g = blockIdx.x * 8 + (threadIdx.x >> 5);  // 0 .. B*H*N-1
    const int b   = g / (12 * 1024);
    const int r   = g % (12 * 1024);
    const int hh  = r >> 10;
    const int tok = r & 1023;
    const int ty  = tok >> 5, tx = tok & 31;

    const float refx = (tx + 0.5f) / 32.f;
    const float refy = (ty + 0.5f) / 32.f;

    const float* qrow = qkv + ((size_t)b * 1024 + tok) * 2304 + hh * 64;
    const float q0 = qrow[lane], q1 = qrow[lane + 32];
    const float* offr = off + ((size_t)b * 1024 + tok) * 96 + hh * 8;

    float gxs[4], gys[4], dots[4];
    #pragma unroll
    for (int p = 0; p < 4; p++) {
        const float lx = 2.f * (refx + offr[p * 2 + 0] * (1.f / 32.f)) - 1.f;
        const float ly = 2.f * (refy + offr[p * 2 + 1] * (1.f / 32.f)) - 1.f;
        const float gx = ((lx + 1.f) * 32.f - 1.f) * 0.5f;
        const float gy = ((ly + 1.f) * 32.f - 1.f) * 0.5f;
        gxs[p] = gx; gys[p] = gy;

        const float fx = floorf(gx), fy = floorf(gy);
        const float wx = gx - fx, wy = gy - fy;
        const int x0 = (int)fx, y0 = (int)fy;
        float sk0 = 0.f, sk1 = 0.f;
        #pragma unroll
        for (int cy = 0; cy < 2; cy++) {
            #pragma unroll
            for (int cx = 0; cx < 2; cx++) {
                const int xi = x0 + cx, yi = y0 + cy;
                if (xi >= 0 && xi < 32 && yi >= 0 && yi < 32) {
                    const float ww = (cx ? wx : 1.f - wx) * (cy ? wy : 1.f - wy);
                    const float* kr = qkv + ((size_t)b * 1024 + yi * 32 + xi) * 2304 + 768 + hh * 64;
                    sk0 += ww * kr[lane];
                    sk1 += ww * kr[lane + 32];
                }
            }
        }
        float d = q0 * sk0 + q1 * sk1;
        #pragma unroll
        for (int m = 16; m >= 1; m >>= 1) d += __shfl_xor(d, m, 32);
        dots[p] = d * 0.125f;
    }

    const float mx = fmaxf(fmaxf(dots[0], dots[1]), fmaxf(dots[2], dots[3]));
    float e[4], se = 0.f;
    #pragma unroll
    for (int p = 0; p < 4; p++) { e[p] = __expf(dots[p] - mx); se += e[p]; }
    const float inv = 1.f / se;

    float o0 = 0.f, o1 = 0.f;
    #pragma unroll
    for (int p = 0; p < 4; p++) {
        const float a = e[p] * inv;
        const float fx = floorf(gxs[p]), fy = floorf(gys[p]);
        const float wx = gxs[p] - fx, wy = gys[p] - fy;
        const int x0 = (int)fx, y0 = (int)fy;
        #pragma unroll
        for (int cy = 0; cy < 2; cy++) {
            #pragma unroll
            for (int cx = 0; cx < 2; cx++) {
                const int xi = x0 + cx, yi = y0 + cy;
                if (xi >= 0 && xi < 32 && yi >= 0 && yi < 32) {
                    const float ww = (cx ? wx : 1.f - wx) * (cy ? wy : 1.f - wy);
                    const float* vr = qkv + ((size_t)b * 1024 + yi * 32 + xi) * 2304 + 1536 + hh * 64;
                    o0 += a * ww * vr[lane];
                    o1 += a * ww * vr[lane + 32];
                }
            }
        }
    }
    __bf16* orow = ao + ((size_t)b * 1024 + tok) * 768 + hh * 64;
    orow[lane]      = (__bf16)o0;
    orow[lane + 32] = (__bf16)o1;
}

// ---------------- host orchestration ----------------
extern "C" void kernel_launch(void* const* d_in, const int* in_sizes, int n_in,
                              void* d_out, int out_size, void* d_ws, size_t ws_size,
                              hipStream_t stream) {
    (void)in_sizes; (void)n_in; (void)out_size; (void)ws_size;

    const float* x       = (const float*)d_in[0];
    const float* patch_w = (const float*)d_in[1];
    const float* patch_b = (const float*)d_in[2];
    const float* qkv_w   = (const float*)d_in[3];
    const float* off_w   = (const float*)d_in[4];
    const float* off_b   = (const float*)d_in[5];
    const float* proj_w  = (const float*)d_in[6];
    const float* proj_b  = (const float*)d_in[7];
    const float* n1_w    = (const float*)d_in[8];
    const float* n1_b    = (const float*)d_in[9];
    const float* n2_w    = (const float*)d_in[10];
    const float* n2_b    = (const float*)d_in[11];
    const float* fc1_w   = (const float*)d_in[12];
    const float* fc1_b   = (const float*)d_in[13];
    const float* fc2_w   = (const float*)d_in[14];
    const float* fc2_b   = (const float*)d_in[15];

    float* h = (float*)d_out;                    // residual stream [2048,768] fp32

    char* ws = (char*)d_ws;
    size_t cur = 0;
    auto alloc = [&](size_t bytes) -> void* {
        void* p = (void*)(ws + cur);
        cur += (bytes + 255) & ~(size_t)255;
        return p;
    };
    __bf16* Apatch = (__bf16*)alloc(2048ull * 768  * 2);
    __bf16* xn     = (__bf16*)alloc(2048ull * 768  * 2);
    float*  qkvb   = (float* )alloc(2048ull * 2304 * 4);
    __bf16* qbf    = (__bf16*)alloc(2048ull * 768  * 2);
    float*  offb   = (float* )alloc(2048ull * 96   * 4);
    __bf16* aob    = (__bf16*)alloc(2048ull * 768  * 2);
    float*  midb   = (float* )alloc(2048ull * 3072 * 4);
    __bf16* midbf  = (__bf16*)alloc(2048ull * 3072 * 2);
    __bf16* wbf    = (__bf16*)alloc(3072ull * 768  * 2);   // padded weight staging

    const int M = 2048;
    auto cvt = [&](const float* src, int ld, __bf16* dst, int rpad, int rval, int cols) {
        cvt_bf16_kernel<<<(rpad * cols + 255) / 256, 256, 0, stream>>>(src, ld, dst, rpad, rval, cols);
    };
    auto gemm = [&](const __bf16* A, int lda, const __bf16* W, const float* bias,
                    float* C, int ldc, int N, int K, int flags) {
        gemm_bf16_kernel<<<dim3(M / BM, (N + BN - 1) / BN), 256, 0, stream>>>(
            A, lda, W, bias, C, ldc, M, N, K, flags);
    };

    // patch embed: h = im2col(x) @ patch_w^T + patch_b
    im2col_kernel<<<(2048 * 768 + 255) / 256, 256, 0, stream>>>(x, Apatch);
    cvt(patch_w, 768, wbf, 768, 768, 768);
    gemm(Apatch, 768, wbf, patch_b, h, 768, 768, 768, 1);

    for (int i = 0; i < 12; i++) {
        ln_kernel<<<256, 256, 0, stream>>>(h, n1_w + i * 768, n1_b + i * 768, xn, M);
        cvt(qkv_w + (size_t)i * 2304 * 768, 768, wbf, 2304, 2304, 768);
        gemm(xn, 768, wbf, nullptr, qkvb, 2304, 2304, 768, 0);

        const bool sample = ((i + 1) % 3) != 0;
        if (sample) {
            // offsets from q (first 768 cols of qkv)
            cvt(qkvb, 2304, qbf, 2048, 2048, 768);
            cvt(off_w + (size_t)i * 96 * 768, 768, wbf, 128, 96, 768);
            gemm(qbf, 768, wbf, off_b + i * 96, offb, 96, 96, 768, 1);
            deform_attn_kernel<<<3072, 256, 0, stream>>>(qkvb, offb, aob);
        } else {
            flash_attn_kernel<<<dim3(24, 16), 128, 0, stream>>>(qkvb, aob);
        }

        // h += ao @ proj_w^T + proj_b
        cvt(proj_w + (size_t)i * 768 * 768, 768, wbf, 768, 768, 768);
        gemm(aob, 768, wbf, proj_b + i * 768, h, 768, 768, 768, 3);

        ln_kernel<<<256, 256, 0, stream>>>(h, n2_w + i * 768, n2_b + i * 768, xn, M);
        cvt(fc1_w + (size_t)i * 3072 * 768, 768, wbf, 3072, 3072, 768);
        gemm(xn, 768, wbf, fc1_b + i * 3072, midb, 3072, 3072, 768, 1);
        gelu_kernel<<<(2048 * 3072 + 255) / 256, 256, 0, stream>>>(midb, midbf, 2048 * 3072);
        cvt(fc2_w + (size_t)i * 768 * 3072, 3072, wbf, 768, 768, 3072);
        gemm(midbf, 3072, wbf, fc2_b + i * 768, h, 768, 768, 3072, 3);
    }
}
